// MaskRCNNC4Predictor_88639535055649
// MI455X (gfx1250) — matrix operand order repaired and checked by reading, not compiled, not run on hardware
//
#include <hip/hip_runtime.h>
#include <hip/hip_bf16.h>

// ---- problem constants (fixed by setup_inputs) ----
#define NB   16      // ROIs
#define CIN  256
#define HH   14
#define WW   14
#define DD   256     // dim_reduced
#define CC   20      // classifiers
#define MM   512     // Nystrom centers
#define QQ   (NB*HH*WW)      // 3136 input pixels
#define FWD  28
#define PP   (NB*FWD*FWD)    // 12544 output pixels

#define TPI  4               // ny tiles staged per sync period (64 centers)

typedef __attribute__((ext_vector_type(16))) _Float16 v16h;
typedef __attribute__((ext_vector_type(8)))  _Float16 v8h;
typedef __attribute__((ext_vector_type(4)))  _Float16 v4h;
typedef __attribute__((ext_vector_type(8)))  float    v8f;
typedef __attribute__((ext_vector_type(4)))  float    v4f;

static __device__ __forceinline__ v16h cat8(v8h a, v8h b) {
  return __builtin_shufflevector(a, b, 0,1,2,3,4,5,6,7,8,9,10,11,12,13,14,15);
}

// ---- pass 1: x [N,Cin,H,W] f32 -> xh [q, c] f16 (c contiguous); base[q] = pixel row base
__global__ void mrc4_prep_x(const float* __restrict__ x, _Float16* __restrict__ xh,
                            int* __restrict__ base) {
  int q = blockIdx.x;            // 0..3135
  int c = threadIdx.x;           // 0..255
  int n = q / (HH*WW), hw = q % (HH*WW);
  xh[q*DD + c] = (_Float16)x[(n*CIN + c)*(HH*WW) + hw];
  if (c == 0) {
    int h = hw / WW, w = hw % WW;
    base[q] = n*(FWD*FWD) + (2*h)*FWD + 2*w;
  }
}

// ---- pass 2: conv_w [Cin,D,2,2] f32 -> whT [ij][o][c] f16 (c contiguous)
__global__ void mrc4_prep_w(const float* __restrict__ cw, _Float16* __restrict__ whT) {
  int ijo = blockIdx.x;          // ij*256 + o
  int c = threadIdx.x;
  int ij = ijo >> 8, o = ijo & 255;
  whT[ijo*CIN + c] = (_Float16)cw[(c*DD + o)*4 + ij];
}

// ---- pass 3: ConvTranspose as 4 GEMMs via WMMA; epilogue bias+ReLU+normalize -> f16 f[p,d]
__global__ void mrc4_conv_wmma(const _Float16* __restrict__ xh, const _Float16* __restrict__ whT,
                               const int* __restrict__ base, const float* __restrict__ bias,
                               const float* __restrict__ mean, const float* __restrict__ mean_norm,
                               _Float16* __restrict__ f) {
  int wid  = blockIdx.x * 8 + (threadIdx.x >> 5);   // 12544 wave-tiles total
  int lane = threadIdx.x & 31;
  int ln = lane & 15, hi = lane >> 4;
  int qt   = wid % (QQ/16);
  int rest = wid / (QQ/16);
  int ot = rest & 15;
  int ij = rest >> 4;
  int q0 = qt*16, o0 = ot*16;

  const _Float16* ap = xh  + (q0 + ln)*DD + hi*8;            // A: pixel rows
  const _Float16* bp = whT + (ij*256 + o0 + ln)*CIN + hi*8;  // B: o columns
  v8f acc = {};
#pragma unroll
  for (int ks = 0; ks < DD/32; ++ks) {
    v8h a0 = *(const v8h*)(ap + ks*32);
    v8h a1 = *(const v8h*)(ap + ks*32 + 16);
    v16h A = cat8(a0, a1);
    v16h B = cat8(*(const v8h*)(bp + ks*32), *(const v8h*)(bp + ks*32 + 16));
    acc = __builtin_amdgcn_wmma_f32_16x16x32_f16(false, A, false, B, (short)0, acc,
                                                 false, false);
  }
  float scale = 20.0f / mean_norm[0];
  int i = ij >> 1, j = ij & 1;
  int o = o0 + ln;
  float mo = mean[o], bo = bias[o];
#pragma unroll
  for (int v = 0; v < 8; ++v) {
    int q = q0 + v + 8*hi;                       // C/D layout: row = v + 8*hi, col = ln
    int p = base[q] + i*FWD + j;
    float val = acc[v] + bo;
    val = val > 0.0f ? val : 0.0f;
    f[p*DD + o] = (_Float16)((val - mo) * scale);
  }
}

// ---- pass 4: ff[p] = sum_d f[p,d]^2
__global__ void mrc4_ff(const _Float16* __restrict__ f, float* __restrict__ ff) {
  int p = blockIdx.x*256 + threadIdx.x;          // 12544
  const v8h* fp = (const v8h*)(f + p*DD);
  float s = 0.0f;
#pragma unroll 4
  for (int k = 0; k < DD/8; ++k) {
    v8h c = fp[k];
#pragma unroll
    for (int e = 0; e < 8; ++e) { float t = (float)c[e]; s += t*t; }
  }
  ff[p] = s;
}

// ---- pass 5: ny -> f16 + nn[c,m] = ||ny||^2
__global__ void mrc4_prep_ny(const float* __restrict__ ny, _Float16* __restrict__ nyh,
                             float* __restrict__ nn) {
  int r = blockIdx.x*256 + threadIdx.x;          // 0..10239 (c*512+m)
  const float* src = ny + (size_t)r*DD;
  _Float16*   dst = nyh + (size_t)r*DD;
  float s = 0.0f;
#pragma unroll 4
  for (int d = 0; d < DD; d += 4) {
    v4f v = *(const v4f*)(src + d);
    s += v[0]*v[0] + v[1]*v[1] + v[2]*v[2] + v[3]*v[3];
    v4h h; h[0] = (_Float16)v[0]; h[1] = (_Float16)v[1];
           h[2] = (_Float16)v[2]; h[3] = (_Float16)v[3];
    *(v4h*)(dst + d) = h;
  }
  nn[r] = s;
}

// ---- pass 6: background plane = -2
__global__ void mrc4_bg(float* __restrict__ out) {
  int t = blockIdx.x*256 + threadIdx.x;          // 12544
  int n = t / (FWD*FWD), r = t % (FWD*FWD);
  out[n*(CC+1)*(FWD*FWD) + r] = -2.0f;
}

// ---- pass 7: FALKON Gaussian-kernel mmv via WMMA + async-to-LDS double buffering.
// block = 16 waves (512 thr); blockIdx.x = 256-pixel strip (49), blockIdx.y = classifier (20).
// Each wave: 16 pixels, full M=512 in 16-center tiles.
//   B (f^T, full K=256) register-resident (64 VGPRs/wave).
//   ny staged into LDS via global_load_async_to_lds_b128 (ASYNCcnt), 4 tiles (32KB) per
//   sync period, double-buffered => 32 WMMAs per wave between barriers.
__global__ void __launch_bounds__(512)
mrc4_falkon(const _Float16* __restrict__ f, const float* __restrict__ ff,
            const _Float16* __restrict__ nyh, const float* __restrict__ nn,
            const float* __restrict__ alpha, const float* __restrict__ sigma,
            float* __restrict__ out) {
  __shared__ __align__(16) _Float16 smA[2][TPI*16*DD];   // 2 x 32KB = 64KB

  int strip = blockIdx.x;                  // 0..48
  int c     = blockIdx.y;                  // 0..19
  int tid   = threadIdx.x;
  int lane  = tid & 31;
  int ln = lane & 15, hi = lane >> 4;
  int p = strip*256 + (tid >> 5)*16 + ln;  // this lane's pixel (column index)

  const _Float16* nyb = nyh + (size_t)c*MM*DD;
  unsigned ldsA0 = (unsigned)(uintptr_t)(&smA[0][0]);
  unsigned ldsA1 = (unsigned)(uintptr_t)(&smA[1][0]);

  // stage chunk 0 (64 rows x 256 halves = 32KB contiguous): 64B (4 x b128) per thread
  {
    const _Float16* g0 = nyb + tid*32;
    unsigned l0 = ldsA0 + tid*64;
#pragma unroll
    for (int u = 0; u < 4; ++u)
      asm volatile("global_load_async_to_lds_b128 %0, %1, off"
                   :: "v"(l0 + u*16), "v"(g0 + u*8) : "memory");
  }

  // B preload: B[k,n] = f[p, k]; lane holds column ln, K = 8*hi..8*hi+7 and 16+8*hi..16+8*hi+7 per k-slice
  const _Float16* fp = f + (size_t)p*DD + hi*8;
  v16h B[DD/32];
#pragma unroll
  for (int ks = 0; ks < DD/32; ++ks) B[ks] = cat8(*(const v8h*)(fp + ks*32), *(const v8h*)(fp + ks*32 + 16));

  float ffp = ff[p];
  float sg  = sigma[0];
  float inv2s2 = 1.0f / (2.0f * sg * sg);
  const float* nnb = nn    + c*MM;
  const float* alb = alpha + c*MM;

  asm volatile("s_wait_asynccnt 0x0" ::: "memory");
  __syncthreads();

  float score = 0.0f;
  for (int ch = 0; ch < MM/(16*TPI); ++ch) {     // 8 sync periods
    int buf = ch & 1;
    // prefetch next 4-tile chunk into the other buffer (async DMA to LDS)
    if (ch + 1 < MM/(16*TPI)) {
      const _Float16* g0 = nyb + (size_t)(ch + 1)*TPI*16*DD + tid*32;
      unsigned l0 = (buf ? ldsA0 : ldsA1) + tid*64;
#pragma unroll
      for (int u = 0; u < 4; ++u)
        asm volatile("global_load_async_to_lds_b128 %0, %1, off"
                     :: "v"(l0 + u*16), "v"(g0 + u*8) : "memory");
    }
    // consume 4 tiles from LDS: A rows = centers, B columns = pixels
#pragma unroll
    for (int t = 0; t < TPI; ++t) {
      int m0 = (ch*TPI + t)*16;
      const _Float16* aB = &smA[buf][t*16*DD] + ln*DD + hi*8;
      v8f acc = {};
#pragma unroll
      for (int ks = 0; ks < DD/32; ++ks) {
        v8h a0 = *(const v8h*)(aB + ks*32);
        v8h a1 = *(const v8h*)(aB + ks*32 + 16);
        v16h A = cat8(a0, a1);
        acc = __builtin_amdgcn_wmma_f32_16x16x32_f16(false, A, false, B[ks], (short)0, acc,
                                                     false, false);
      }
      // epilogue: lane's centers are m0 + v + 8*hi, v = 0..7
      v4f nl = *(const v4f*)(nnb + m0 + 8*hi);
      v4f nh = *(const v4f*)(nnb + m0 + 8*hi + 4);
      v4f al = *(const v4f*)(alb + m0 + 8*hi);
      v4f ah = *(const v4f*)(alb + m0 + 8*hi + 4);
#pragma unroll
      for (int v = 0; v < 8; ++v) {
        float nnv = (v < 4) ? nl[v] : nh[v-4];
        float alv = (v < 4) ? al[v] : ah[v-4];
        float dsq = ffp + nnv - 2.0f*acc[v];
        score += alv * __expf(-dsq * inv2s2);
      }
    }
    asm volatile("s_wait_asynccnt 0x0" ::: "memory");
    __syncthreads();
  }

  // lanes l / l+16 hold complementary center halves for the same pixel
  score += __shfl_xor(score, 16, 32);
  if (hi == 0) {
    int n = p / (FWD*FWD), r = p % (FWD*FWD);
    out[(n*(CC+1) + (c+1))*(FWD*FWD) + r] = score;
  }
}

extern "C" void kernel_launch(void* const* d_in, const int* in_sizes, int n_in,
                              void* d_out, int out_size, void* d_ws, size_t ws_size,
                              hipStream_t stream) {
  const float* x         = (const float*)d_in[0];
  const float* conv_w    = (const float*)d_in[1];
  const float* conv_b    = (const float*)d_in[2];
  const float* mean      = (const float*)d_in[3];
  const float* mean_norm = (const float*)d_in[4];
  const float* ny        = (const float*)d_in[5];
  const float* alpha     = (const float*)d_in[6];
  const float* sigma     = (const float*)d_in[7];
  float* out = (float*)d_out;

  // workspace layout (256B aligned)
  char* w = (char*)d_ws;
  size_t off = 0;
  auto take = [&](size_t bytes) { void* p = w + off; off = (off + bytes + 255) & ~(size_t)255; return p; };
  _Float16* xh   = (_Float16*)take((size_t)QQ*DD*2);      // 1.6 MB
  int*      base = (int*)     take((size_t)QQ*4);
  _Float16* whT  = (_Float16*)take((size_t)4*DD*CIN*2);   // 0.5 MB
  _Float16* ff16 = (_Float16*)take((size_t)PP*DD*2);      // 6.4 MB (features f)
  float*    ffsq = (float*)   take((size_t)PP*4);
  _Float16* nyh  = (_Float16*)take((size_t)CC*MM*DD*2);   // 5.2 MB
  float*    nnsq = (float*)   take((size_t)CC*MM*4);

  mrc4_prep_x  <<<QQ, 256, 0, stream>>>(x, xh, base);
  mrc4_prep_w  <<<4*DD, 256, 0, stream>>>(conv_w, whT);
  mrc4_conv_wmma<<<(QQ/16)*16*4/8, 256, 0, stream>>>(xh, whT, base, conv_b, mean,
                                                     mean_norm, ff16);
  mrc4_ff      <<<PP/256, 256, 0, stream>>>(ff16, ffsq);
  mrc4_prep_ny <<<CC*MM/256, 256, 0, stream>>>(ny, nyh, nnsq);
  mrc4_bg      <<<PP/256, 256, 0, stream>>>(out);
  mrc4_falkon  <<<dim3(PP/256, CC), 512, 0, stream>>>(ff16, ffsq, nyh, nnsq, alpha,
                                                      sigma, out);
}
